// AxialAttentionBlock_69561290326112
// MI455X (gfx1250) — matrix operand order repaired and checked by reading, not compiled, not run on hardware
//
#include <hip/hip_runtime.h>

typedef _Float16 half_t;
typedef __attribute__((ext_vector_type(16))) _Float16 v16h;
typedef __attribute__((ext_vector_type(8)))  _Float16 v8h;
typedef __attribute__((ext_vector_type(8)))  float    v8f;

#define B_    4
#define C_    512
#define HW_   128
#define HEADS 8
#define DH    64
#define L_    128
#define INNER 512
#define SCALE 0.125f

// LDS strides (elements), padded to avoid bank conflicts
#define XS_STRIDE 520   // half, xs  [L][C]
#define QK_STRIDE 72    // half, q,k [L][DH]
#define V_STRIDE  136   // half, v   [DH][L]
#define P_STRIDE  136   // half, p   [L][L]
#define S_STRIDE  132   // float, s  [L][L]

// LDS byte offsets (all multiples of 16)
#define XS_OFF 0
#define Q_OFF  (XS_OFF + L_*XS_STRIDE*2)
#define K_OFF  (Q_OFF  + L_*QK_STRIDE*2)
#define V_OFF  (K_OFF  + L_*QK_STRIDE*2)
#define P_OFF  (V_OFF  + DH*V_STRIDE*2)
#define S_OFF  (P_OFF  + L_*P_STRIDE*2)
#define ATTN_LDS_BYTES (S_OFF + L_*S_STRIDE*4)   // 289792 B < 320 KB

#define PROJ_ST_STRIDE 520
#define PROJ_LDS_BYTES (L_*PROJ_ST_STRIDE*2)     // 133120 B

// ---- CDNA5 async global->LDS copy (ASYNCcnt path) -------------------------
// LDS byte offset = low 32 bits of the generic pointer (flat LDS aperture:
// addr[31:0] is the LDS address). 16 bytes per lane, per-lane addresses.
__device__ inline void async_copy16(const half_t* gsrc, half_t* ldst) {
  unsigned l = (unsigned)(size_t)ldst;
  unsigned long long ga = (unsigned long long)(size_t)gsrc;
  asm volatile("global_load_async_to_lds_b128 %0, %1, off"
               :: "v"(l), "v"(ga) : "memory");
}
__device__ inline void async_wait0() {
  asm volatile("s_wait_asynccnt 0x0" ::: "memory");
}

// ---- WMMA operand helpers -------------------------------------------------
// 16-bit A-matrix 16x32 layout: lane holds row m=lane&15; g=lane>>4.
// Halves t=0..15 -> K = kbase+16g+t (contiguous, matches loadB).
__device__ inline v16h cat8(v8h lo, v8h hi) {
  return __builtin_shufflevector(lo, hi, 0,1,2,3,4,5,6,7,8,9,10,11,12,13,14,15);
}
__device__ inline v16h loadA(const half_t* p /* = row + kbase + 16*g */) {
  v8h lo = *(const v8h*)(p);
  v8h hi = *(const v8h*)(p + 8);
  return cat8(lo, hi);
}
// 16-bit B-matrix 32x16: lane holds column n=lane&15; halves t=0..15 ->
// K = kbase + 16*(lane>>4) + t : one contiguous 32-byte run.
__device__ inline v16h loadB(const half_t* p /* = col + kbase + 16*g */) {
  v8h lo = *(const v8h*)(p);
  v8h hi = *(const v8h*)(p + 8);
  return cat8(lo, hi);
}
__device__ inline v8h to_h8(v8f c) {
  v8h r;
  #pragma unroll
  for (int i = 0; i < 8; ++i) r[i] = (half_t)c[i];
  return r;
}
__device__ inline v8f wmma16(v16h a, v16h b, v8f c) {
  return __builtin_amdgcn_wmma_f32_16x16x32_f16(false, a, false, b,
                                                (short)0, c, false, false);
}

// ---- Attention kernel: one workgroup (8 waves) per 1-D instance -----------
// PASS 0: read x (f32, column gather along H, scalar stage).
// PASS 1: read hbuf (f16, layout [b][hh][i][c], async b128 stage).
// out_buf layout: [inst][i][o] (f16).
template<int PASS>
__global__ void __launch_bounds__(256, 1)
attn_kernel(const float* __restrict__ x, const half_t* __restrict__ hbuf,
            const half_t* __restrict__ wqkv, half_t* __restrict__ out_buf) {
  extern __shared__ __attribute__((aligned(16))) char smem[];
  half_t* xs  = (half_t*)(smem + XS_OFF);   // [L][C]   xs^T
  half_t* qld = (half_t*)(smem + Q_OFF);    // [L][DH]  q[i][dd]
  half_t* kld = (half_t*)(smem + K_OFF);    // [L][DH]  k[j][dd]
  half_t* vld = (half_t*)(smem + V_OFF);    // [DH][L]  v[dd][j]
  half_t* pld = (half_t*)(smem + P_OFF);    // [L][L]   p[i][j]
  float*  sws = (float*)(smem + S_OFF);     // [L][L]   scores f32

  const int inst = blockIdx.x;
  const int tid  = threadIdx.x;
  const int wv   = tid >> 5;
  const int ln   = tid & 31;
  const int lh   = ln & 15;
  const int g    = ln >> 4;
  const int b    = inst >> 7;
  const int col  = inst & 127;    // w for PASS0, h for PASS1

  // Phase 0: stage xs^T into LDS as f16
  if (PASS == 0) {
    for (int j = 0; j < 256; ++j) {
      int flat = tid + 256 * j;
      int c = flat >> 7;
      int i = flat & 127;
      float v = x[(((size_t)(b * C_ + c)) * HW_ + i) * HW_ + col];
      xs[i * XS_STRIDE + c] = (half_t)v;
    }
  } else {
    // hbuf[inst] is [i][c] contiguous: pure async b128 copy into padded rows
    const half_t* src = hbuf + (size_t)inst * L_ * C_;
    for (int j = 0; j < 32; ++j) {
      int flat8 = (tid + 256 * j) * 8;
      int i = flat8 >> 9;
      int c = flat8 & 511;
      async_copy16(src + flat8, xs + i * XS_STRIDE + c);
    }
    async_wait0();
  }
  __syncthreads();

  const v8f vzero = {};

  for (int h = 0; h < HEADS; ++h) {
    // ---- QKV: combined M=192 (q|k|v), N=L=128, K=C=512; n-tile = wave ----
    for (int mt = 0; mt < 12; ++mt) {
      int sec    = mt >> 2;                 // 0=q, 1=k, 2=v
      int within = (mt & 3) * 16 + lh;
      const half_t* arow = wqkv + ((size_t)(sec * INNER + h * DH + within)) * C_;
      const int i = wv * 16 + lh;
      const half_t* bcol = xs + i * XS_STRIDE;
      v8f acc = vzero;
      for (int kk = 0; kk < 16; ++kk) {
        v16h a  = loadA(arow + kk * 32 + 16 * g);
        v16h bb = loadB(bcol + kk * 32 + 16 * g);
        acc = wmma16(a, bb, acc);
      }
      if (sec == 0) {
        *(v8h*)(qld + i * QK_STRIDE + (mt & 3) * 16 + 8 * g) = to_h8(acc);
      } else if (sec == 1) {
        *(v8h*)(kld + i * QK_STRIDE + (mt & 3) * 16 + 8 * g) = to_h8(acc);
      } else {
        #pragma unroll
        for (int v2 = 0; v2 < 8; ++v2)
          vld[((mt & 3) * 16 + 8 * g + v2) * V_STRIDE + i] = (half_t)acc[v2];
      }
    }
    __syncthreads();

    // ---- S = Q K^T : wave handles rows [16w,16w+16), K-dim = 64 ----------
    {
      const half_t* arow = qld + (wv * 16 + lh) * QK_STRIDE;
      for (int nt = 0; nt < 8; ++nt) {
        const half_t* bcol = kld + (nt * 16 + lh) * QK_STRIDE;
        v8f acc = vzero;
        #pragma unroll
        for (int kk = 0; kk < 2; ++kk) {
          v16h a  = loadA(arow + kk * 32 + 16 * g);
          v16h bb = loadB(bcol + kk * 32 + 16 * g);
          acc = wmma16(a, bb, acc);
        }
        #pragma unroll
        for (int v2 = 0; v2 < 8; ++v2)
          sws[(wv * 16 + 8 * g + v2) * S_STRIDE + nt * 16 + lh] = acc[v2] * SCALE;
      }
    }

    // ---- softmax over the wave's 16 rows (wave-local) --------------------
    for (int r = 0; r < 16; ++r) {
      int row = wv * 16 + r;
      float vals[4];
      float mx = -3.0e38f;
      #pragma unroll
      for (int cc = 0; cc < 4; ++cc) {
        vals[cc] = sws[row * S_STRIDE + ln + 32 * cc];
        mx = fmaxf(mx, vals[cc]);
      }
      #pragma unroll
      for (int off = 16; off > 0; off >>= 1)
        mx = fmaxf(mx, __shfl_xor(mx, off, 32));
      float sum = 0.f;
      #pragma unroll
      for (int cc = 0; cc < 4; ++cc) {
        vals[cc] = __expf(vals[cc] - mx);
        sum += vals[cc];
      }
      #pragma unroll
      for (int off = 16; off > 0; off >>= 1)
        sum += __shfl_xor(sum, off, 32);
      float inv = 1.0f / sum;
      #pragma unroll
      for (int cc = 0; cc < 4; ++cc)
        pld[row * P_STRIDE + ln + 32 * cc] = (half_t)(vals[cc] * inv);
    }

    // ---- O = P V : M = wave rows, N = 64 (4 tiles), K = 128 (4 steps) ----
    {
      const half_t* arow = pld + (wv * 16 + lh) * P_STRIDE;
      for (int nt = 0; nt < 4; ++nt) {
        const half_t* bcol = vld + (nt * 16 + lh) * V_STRIDE;
        v8f acc = vzero;
        #pragma unroll
        for (int kk = 0; kk < 4; ++kk) {
          v16h a  = loadA(arow + kk * 32 + 16 * g);
          v16h bb = loadB(bcol + kk * 32 + 16 * g);
          acc = wmma16(a, bb, acc);
        }
        int o = h * DH + nt * 16 + lh;
        // out_buf layout [inst][i][o]; lanes of one v2 step are o-contiguous
        #pragma unroll
        for (int v2 = 0; v2 < 8; ++v2)
          out_buf[((size_t)inst * L_ + wv * 16 + 8 * g + v2) * INNER + o] =
              (half_t)acc[v2];
      }
    }
    __syncthreads();   // q/k/v reused next head
  }
}

// ---- Projection kernel: y = Wproj @ O, fused residual on final pass ------
// out_buf [inst][i][o] is staged by async b128 copies into LDS [i][o].
template<int PASS>
__global__ void __launch_bounds__(256, 1)
proj_kernel(const half_t* __restrict__ wproj, const half_t* __restrict__ out_buf,
            const float* __restrict__ x, half_t* __restrict__ hbuf,
            float* __restrict__ out) {
  extern __shared__ __attribute__((aligned(16))) char smem[];
  half_t* st = (half_t*)smem;     // [i=128][o=512], stride PROJ_ST_STRIDE

  const int inst = blockIdx.x;
  const int tid  = threadIdx.x;
  const int wv   = tid >> 5;
  const int ln   = tid & 31;
  const int lh   = ln & 15;
  const int g    = ln >> 4;
  const int b    = inst >> 7;
  const int col  = inst & 127;

  const half_t* src = out_buf + (size_t)inst * L_ * INNER;
  for (int j = 0; j < 32; ++j) {
    int flat8 = (tid + 256 * j) * 8;
    int i = flat8 >> 9;
    int o = flat8 & 511;
    async_copy16(src + flat8, st + i * PROJ_ST_STRIDE + o);
  }
  async_wait0();
  __syncthreads();

  const v8f vzero = {};
  for (int r = 0; r < 4; ++r) {
    int mt = wv + 8 * r;
    const half_t* arow = wproj + (size_t)(mt * 16 + lh) * INNER;
    // prefetch next m-tile's weight row into caches
    int mt_n = (r < 3) ? (mt + 8) : mt;
    __builtin_prefetch(wproj + (size_t)(mt_n * 16 + lh) * INNER, 0, 1);
    v8f acc[8];
    #pragma unroll
    for (int nt = 0; nt < 8; ++nt) acc[nt] = vzero;
    for (int kk = 0; kk < 16; ++kk) {
      v16h a = loadA(arow + kk * 32 + 16 * g);
      #pragma unroll
      for (int nt = 0; nt < 8; ++nt) {
        v16h bb = loadB(st + (nt * 16 + lh) * PROJ_ST_STRIDE + kk * 32 + 16 * g);
        acc[nt] = wmma16(a, bb, acc[nt]);
      }
    }
    #pragma unroll
    for (int nt = 0; nt < 8; ++nt) {
      int i = nt * 16 + lh;
      if (PASS == 0) {
        // hbuf layout [b][hh][i][c]: c = mt*16+8g+v2 contiguous -> packed b128
        size_t idx = (((size_t)(b * HW_ + col)) * HW_ + i) * C_ + mt * 16 + 8 * g;
        *(v8h*)(hbuf + idx) = to_h8(acc[nt]);
      } else {
        #pragma unroll
        for (int v2 = 0; v2 < 8; ++v2) {
          int c = mt * 16 + 8 * g + v2;
          size_t idx = (((size_t)(b * C_ + c)) * HW_ + col) * HW_ + i;
          out[idx] = x[idx] + acc[nt][v2];
        }
      }
    }
  }
}

// ---- weight f32 -> f16 conversion ----------------------------------------
__global__ void convert_weights(const float* __restrict__ wqkv,
                                const float* __restrict__ wproj,
                                half_t* __restrict__ wqkv_h,
                                half_t* __restrict__ wproj_h) {
  int t = blockIdx.x * blockDim.x + threadIdx.x;
  const int n1 = 3 * INNER * C_;
  const int n2 = C_ * INNER;
  if (t < n1) wqkv_h[t] = (half_t)wqkv[t];
  if (t < n2) wproj_h[t] = (half_t)wproj[t];
}

extern "C" void kernel_launch(void* const* d_in, const int* in_sizes, int n_in,
                              void* d_out, int out_size, void* d_ws, size_t ws_size,
                              hipStream_t stream) {
  const float* x     = (const float*)d_in[0];
  const float* wqkv  = (const float*)d_in[1];
  const float* wproj = (const float*)d_in[2];
  float* out = (float*)d_out;
  char*  ws  = (char*)d_ws;

  half_t* wqkv_h  = (half_t*)(ws);                           // 1,572,864 B
  half_t* wproj_h = (half_t*)(ws + 1572864);                 //   524,288 B
  half_t* out_buf = (half_t*)(ws + 2097152);                 // 67,108,864 B
  half_t* hbuf    = (half_t*)(ws + 2097152 + 67108864);      // 67,108,864 B

  convert_weights<<<(3 * INNER * C_ + 255) / 256, 256, 0, stream>>>(
      wqkv, wproj, wqkv_h, wproj_h);

  const int NINST = B_ * HW_;  // 512 instances per pass
  attn_kernel<0><<<NINST, 256, ATTN_LDS_BYTES, stream>>>(x, nullptr, wqkv_h, out_buf);
  proj_kernel<0><<<NINST, 256, PROJ_LDS_BYTES, stream>>>(wproj_h, out_buf, nullptr, hbuf, nullptr);
  attn_kernel<1><<<NINST, 256, ATTN_LDS_BYTES, stream>>>(nullptr, hbuf, wqkv_h, out_buf);
  proj_kernel<1><<<NINST, 256, PROJ_LDS_BYTES, stream>>>(wproj_h, out_buf, x, nullptr, out);
}
